// PrototypeBank_67284957659354
// MI455X (gfx1250) — matrix operand order repaired and checked by reading, not compiled, not run on hardware
//
#include <hip/hip_runtime.h>
#include <hip/hip_bf16.h>
#include <math.h>

// ---------------------------------------------------------------------------
// PrototypeBank.assign for MI455X (gfx1250, wave32, WMMA bf16 16x16x32)
//   q = softmax( (l2norm(emb) @ l2norm(protos)^T) / temperature )
// N=65536, P=4096, D=768 (divisible by all tile sizes used here).
// ---------------------------------------------------------------------------

typedef __attribute__((ext_vector_type(16))) __bf16 v16bf;
typedef __attribute__((ext_vector_type(8)))  __bf16 v8bf;
typedef __attribute__((ext_vector_type(8)))  float  v8f;

static constexpr int Dc      = 768;   // feature dim (compile-time)
static constexpr int ROWS    = 16;    // emb rows per workgroup (one WMMA M tile)
static constexpr int AS      = Dc + 8;   // LDS A-tile stride in bf16 (bank-conflict pad)

__device__ __forceinline__ float wave_sum(float v) {
#pragma unroll
  for (int o = 16; o > 0; o >>= 1) v += __shfl_xor(v, o, 32);
  return v;
}
__device__ __forceinline__ float wave_max(float v) {
#pragma unroll
  for (int o = 16; o > 0; o >>= 1) v = fmaxf(v, __shfl_xor(v, o, 32));
  return v;
}

// ---------------------------------------------------------------------------
// Kernel A: L2-normalize prototypes, f32 -> bf16, row-major (P x D) into ws.
// One wave per row; 8 rows per 256-thread block.
// ---------------------------------------------------------------------------
__global__ void proto_norm_bf16_kernel(const float* __restrict__ proto,
                                       __bf16* __restrict__ out, int P) {
  const int wave = threadIdx.x >> 5;
  const int lane = threadIdx.x & 31;
  const int row  = blockIdx.x * 8 + wave;
  if (row >= P) return;

  const float* src = proto + (size_t)row * Dc;
  float v[Dc / 32];
  float ss = 0.f;
#pragma unroll
  for (int j = 0; j < Dc / 32; ++j) {
    v[j] = src[lane + 32 * j];
    ss += v[j] * v[j];
  }
  ss = wave_sum(ss);
  const float scale = 1.0f / fmaxf(sqrtf(ss), 1e-12f);

  __bf16* dst = out + (size_t)row * Dc;
#pragma unroll
  for (int j = 0; j < Dc / 32; ++j)
    dst[lane + 32 * j] = (__bf16)(v[j] * scale);
}

// ---------------------------------------------------------------------------
// Kernel B: per workgroup = 16 emb rows x full P columns.
//   stage 1: normalize 16 emb rows -> bf16 A-tile in LDS
//   stage 2: WMMA bf16 16x16x32 over K=768; logits*(1/T) -> LDS (f32)
//   stage 3: row softmax (max-sub), write q to global
// blockDim = 128 (4 waves). Dynamic LDS = A-tile + logits buffer (~281 KB,
// within CDNA5's 320 KB/WGP).
// ---------------------------------------------------------------------------
__global__ void __launch_bounds__(128)
assign_softmax_kernel(const float* __restrict__ emb,
                      const __bf16* __restrict__ protos_bf16,
                      const float* __restrict__ temperature,
                      float* __restrict__ out, int P) {
  extern __shared__ char smem_raw[];
  __bf16* Abuf  = (__bf16*)smem_raw;                       // [ROWS][AS]
  float* logits = (float*)(smem_raw + ROWS * AS * sizeof(__bf16)); // [ROWS][LP]
  const int LP = P + 4;                                    // f32 stride pad

  const int wave  = threadIdx.x >> 5;   // 0..3
  const int lane  = threadIdx.x & 31;
  const int lhalf = lane >> 4;          // 0: lanes 0-15, 1: lanes 16-31
  const int lidx  = lane & 15;

  const float invT = 1.0f / temperature[0];
  const int row0 = blockIdx.x * ROWS;

  // ---- stage 1: normalize this block's 16 emb rows into LDS (bf16) --------
#pragma unroll
  for (int i = 0; i < 4; ++i) {
    const int r = wave * 4 + i;                 // 0..15
    const float* src = emb + (size_t)(row0 + r) * Dc;
    float v[Dc / 32];
    float ss = 0.f;
#pragma unroll
    for (int j = 0; j < Dc / 32; ++j) {
      v[j] = src[lane + 32 * j];
      ss += v[j] * v[j];
    }
    ss = wave_sum(ss);
    const float scale = 1.0f / fmaxf(sqrtf(ss), 1e-12f);
    __bf16* dst = Abuf + r * AS;
#pragma unroll
    for (int j = 0; j < Dc / 32; ++j)
      dst[lane + 32 * j] = (__bf16)(v[j] * scale);
  }
  __syncthreads();

  // ---- stage 2: GEMM via v_wmma_f32_16x16x32_bf16 -------------------------
  // A fragment (16x32 bf16): lanes 0-15 hold row M=lane, K 0..15 contiguous;
  // lanes 16-31 hold row M=lane-16, K 16..31 contiguous  -> two 16B chunks.
  // B fragment (32x16 bf16): lane<16 holds col N=lane, K 0..15 contiguous;
  // lane>=16 holds col N=lane-16, K 16..31  -> one 32B contiguous chunk of a
  // row-major prototype row (since B[k][n] = c[n][k]).
  const int tiles_per_wave = P / (16 * 4);
  const __bf16* arow = Abuf + lidx * AS + lhalf * 16;

  for (int tt = 0; tt < tiles_per_wave; ++tt) {
    const int t  = wave * tiles_per_wave + tt;
    const int n0 = t * 16;
    const __bf16* brow = protos_bf16 + (size_t)(n0 + lidx) * Dc + lhalf * 16;

    v8f acc = {};
#pragma unroll
    for (int k0 = 0; k0 < Dc; k0 += 32) {
      v8bf a0 = *(const v8bf*)(arow + k0);        // K chunk 0 (16 B, LDS)
      v8bf a1 = *(const v8bf*)(arow + k0 + 8);    // K chunk 1 (16 B, LDS)
      v16bf a;
#pragma unroll
      for (int e = 0; e < 8; ++e) { a[e] = a0[e]; a[e + 8] = a1[e]; }
      v16bf b = *(const v16bf*)(brow + k0);       // 32 B contiguous (global/L2)
      acc = __builtin_amdgcn_wmma_f32_16x16x32_bf16(
          /*neg_a=*/false, a, /*neg_b=*/false, b,
          /*c_mod=*/(short)0, acc, /*reuse_a=*/false, /*reuse_b=*/false);
    }
    // C/D layout: VGPR r, lanes 0-15 -> M=r, lanes 16-31 -> M=r+8; N=lidx.
#pragma unroll
    for (int r = 0; r < 8; ++r)
      logits[(r + 8 * lhalf) * LP + n0 + lidx] = acc[r] * invT;
  }
  __syncthreads();

  // ---- stage 3: row softmax + store ---------------------------------------
#pragma unroll
  for (int i = 0; i < 4; ++i) {
    const int m = wave * 4 + i;
    float* Lrow = logits + m * LP;

    float mx = -3.402823466e+38f;
    for (int c = lane; c < P; c += 32) mx = fmaxf(mx, Lrow[c]);
    mx = wave_max(mx);

    float s = 0.f;
    for (int c = lane; c < P; c += 32) {
      const float e = __expf(Lrow[c] - mx);
      Lrow[c] = e;          // stash exp back to LDS
      s += e;
    }
    s = wave_sum(s);
    const float inv = 1.0f / s;

    float* orow = out + (size_t)(row0 + m) * P;
    for (int c = lane; c < P; c += 32) orow[c] = Lrow[c] * inv;
  }
}

// ---------------------------------------------------------------------------
extern "C" void kernel_launch(void* const* d_in, const int* in_sizes, int n_in,
                              void* d_out, int out_size, void* d_ws, size_t ws_size,
                              hipStream_t stream) {
  const float* emb   = (const float*)d_in[0];
  const float* proto = (const float*)d_in[1];
  const float* temp  = (const float*)d_in[2];
  float* out = (float*)d_out;

  const int N = in_sizes[0] / Dc;   // 65536
  const int P = in_sizes[1] / Dc;   // 4096

  __bf16* protos_bf16 = (__bf16*)d_ws;   // P*Dc*2 bytes (~6 MB)

  // Kernel A: normalize prototypes to bf16
  proto_norm_bf16_kernel<<<(P + 7) / 8, 256, 0, stream>>>(proto, protos_bf16, P);

  // Kernel B: fused normalize + WMMA GEMM + softmax
  const int LP = P + 4;
  const size_t smem = (size_t)ROWS * AS * sizeof(__bf16) +
                      (size_t)ROWS * LP * sizeof(float);   // ~281 KB < 320 KB/WGP
  hipFuncSetAttribute((const void*)assign_softmax_kernel,
                      hipFuncAttributeMaxDynamicSharedMemorySize, (int)smem);
  assign_softmax_kernel<<<N / ROWS, 128, smem, stream>>>(
      emb, protos_bf16, temp, out, P);
}
